// MACCLLoss_10746008174774
// MI455X (gfx1250) — matrix operand order repaired and checked by reading, not compiled, not run on hardware
//
#include <hip/hip_runtime.h>
#include <hip/hip_bf16.h>

// ---------------- problem constants ----------------
#define BN 8192
#define DN 256
#define INV_TEMP (1.0f / 0.07f)
#define NSPLIT 8                      // column-sweep split for machine fill

typedef __bf16 v16bf __attribute__((ext_vector_type(16)));
typedef __bf16 v8bf  __attribute__((ext_vector_type(8)));
typedef float  v8f   __attribute__((ext_vector_type(8)));

// ======================================================================
// Kernel 0: zero scalar accumulators, pos/neg atomic buffers, output.
// ======================================================================
__global__ void maccl_zero_kernel(float* __restrict__ scal,
                                  float* __restrict__ pos,
                                  float* __restrict__ neg,
                                  float* __restrict__ out) {
    const int i = blockIdx.x * 256 + threadIdx.x;
    if (i < BN) { pos[i] = 0.0f; neg[i] = 0.0f; }
    if (i < 4)  scal[i] = 0.0f;
    if (i == 0) out[0] = 0.0f;
}

// ======================================================================
// Kernel 1: per-row prep.  One block (256 thr) per row, one thread per dim.
//  - dist_sq[row] = ||x - center||^2
//  - masked Sum(x), Sum(x^2), count  (atomics -> scal[0..2])
//  - split-bf16 normalized features: hi = bf16(x/||x||), lo = bf16(residual)
// ======================================================================
__global__ void maccl_prep_kernel(const float* __restrict__ feats,
                                  const int*   __restrict__ labels,
                                  const float* __restrict__ center,
                                  __bf16* __restrict__ hi,
                                  __bf16* __restrict__ lo,
                                  float*  __restrict__ distSq,
                                  float*  __restrict__ scal) {
    const int row  = blockIdx.x;
    const int tid  = threadIdx.x;
    const int lane = tid & 31;
    const int w    = tid >> 5;

    __shared__ float sD2[8], sX2[8], sX[8];
    __shared__ float bcNormSq;

    const float x  = feats[row * DN + tid];
    const float df = x - center[tid];
    float rD2 = df * df;
    float rX2 = x * x;
    float rX  = x;
#pragma unroll
    for (int o = 16; o > 0; o >>= 1) {
        rD2 += __shfl_down(rD2, o, 32);
        rX2 += __shfl_down(rX2, o, 32);
        rX  += __shfl_down(rX,  o, 32);
    }
    if (lane == 0) { sD2[w] = rD2; sX2[w] = rX2; sX[w] = rX; }
    __syncthreads();
    if (tid == 0) {
        float a = 0.f, b = 0.f, c = 0.f;
#pragma unroll
        for (int i = 0; i < 8; ++i) { a += sD2[i]; b += sX2[i]; c += sX[i]; }
        distSq[row] = a;
        if (labels[row] == 0) {
            atomicAdd(scal + 0, c);   // masked sum(x)
            atomicAdd(scal + 1, b);   // masked sum(x^2)
            atomicAdd(scal + 2, 1.f); // n_normal
        }
        bcNormSq = b;
    }
    __syncthreads();

    const float nrm = fmaxf(sqrtf(bcNormSq), 1e-12f);
    const float fx  = x / nrm;
    const __bf16 h  = (__bf16)fx;
    const float res = fx - (float)h;
    const __bf16 l  = (__bf16)res;
    hi[row * DN + tid] = h;
    lo[row * DN + tid] = l;
}

// ======================================================================
// Kernel 2: fused Gram + contrastive row sums, split-bf16 emulated fp32.
//  grid = (64, NSPLIT) blocks x 256 threads (8 waves).  Wave w owns rows
//  [blockIdx.x*128 + w*16, +16); block sweeps columns
//  [blockIdx.y*(BN/NSPLIT), +BN/NSPLIT) two 16-col tiles at a time.
//  Per iteration: 48 v_wmma_f32_16x16x32_bf16 on TWO independent
//  accumulator chains (hi*hi + hi*lo + lo*hi over K=256), so one chain's
//  loads overlap the other chain's matrix ops.  Epilogue: exp + label
//  masking + diag exclusion in registers; atomicAdd per row at the end.
// ======================================================================
__global__ __launch_bounds__(256, 1)
void maccl_gram_kernel(const __bf16* __restrict__ hi,
                       const __bf16* __restrict__ lo,
                       const int*    __restrict__ labels,
                       float* __restrict__ posOut,
                       float* __restrict__ negOut) {
    const int lane    = threadIdx.x & 31;
    const int wave    = threadIdx.x >> 5;
    const int laneHi  = lane >> 4;   // 0: lanes 0-15, 1: lanes 16-31
    const int lane15  = lane & 15;
    const int rowBase = blockIdx.x * 128 + wave * 16;
    const int colBeg  = blockIdx.y * (BN / NSPLIT);
    const int colEnd  = colBeg + (BN / NSPLIT);
    const int khalfB  = laneHi ? 16 : 0;

    // ---- preload A fragments (this wave's 16 rows, hi and lo planes) ----
    // A operand in this code's contiguous k order: lane m holds row m; lanes 0-15 get K {0..15},
    // lanes 16-31 get K {16..31} within each 32-wide K window.
    const __bf16* aHiRow = hi + (rowBase + lane15) * DN + (laneHi ? 16 : 0);
    const __bf16* aLoRow = lo + (rowBase + lane15) * DN + (laneHi ? 16 : 0);
    v16bf ah[8], al[8];
#pragma unroll
    for (int f = 0; f < 8; ++f) {
        const v8bf h0 = *(const v8bf*)(aHiRow + 32 * f);
        const v8bf h1 = *(const v8bf*)(aHiRow + 32 * f + 8);
        const v8bf l0 = *(const v8bf*)(aLoRow + 32 * f);
        const v8bf l1 = *(const v8bf*)(aLoRow + 32 * f + 8);
#pragma unroll
        for (int i = 0; i < 8; ++i) {
            ah[f][i] = h0[i]; ah[f][i + 8] = h1[i];
            al[f][i] = l0[i]; al[f][i + 8] = l1[i];
        }
    }
    // row labels for this lane's output rows (C layout: VGPR r -> M=r(+8))
    int rl[8];
#pragma unroll
    for (int r = 0; r < 8; ++r) rl[r] = labels[rowBase + r + laneHi * 8];

    float pacc[8] = {0.f, 0.f, 0.f, 0.f, 0.f, 0.f, 0.f, 0.f};
    float nacc[8] = {0.f, 0.f, 0.f, 0.f, 0.f, 0.f, 0.f, 0.f};

    for (int colBase = colBeg; colBase < colEnd; colBase += 32) {
        const int col0 = colBase + lane15;
        const int col1 = col0 + 16;
        const int cl0  = labels[col0];
        const int cl1  = labels[col1];
        // B operand, same contiguous k order as A: lane n holds column n, K 0..15 (lanes 0-15) or
        // K 16..31 (lanes 16-31) -> one contiguous 32-byte load per frag.
        const __bf16* b0h = hi + col0 * DN + khalfB;
        const __bf16* b0l = lo + col0 * DN + khalfB;
        const __bf16* b1h = hi + col1 * DN + khalfB;
        const __bf16* b1l = lo + col1 * DN + khalfB;

        v8f acc0 = {0.f, 0.f, 0.f, 0.f, 0.f, 0.f, 0.f, 0.f};
        v8f acc1 = {0.f, 0.f, 0.f, 0.f, 0.f, 0.f, 0.f, 0.f};
#pragma unroll
        for (int f = 0; f < 8; ++f) {
            const v16bf bh0 = *(const v16bf*)(b0h + 32 * f);
            const v16bf bl0 = *(const v16bf*)(b0l + 32 * f);
            const v16bf bh1 = *(const v16bf*)(b1h + 32 * f);
            const v16bf bl1 = *(const v16bf*)(b1l + 32 * f);
            acc0 = __builtin_amdgcn_wmma_f32_16x16x32_bf16(
                       false, ah[f], false, bh0, (short)0, acc0, false, false);
            acc1 = __builtin_amdgcn_wmma_f32_16x16x32_bf16(
                       false, ah[f], false, bh1, (short)0, acc1, false, false);
            acc0 = __builtin_amdgcn_wmma_f32_16x16x32_bf16(
                       false, ah[f], false, bl0, (short)0, acc0, false, false);
            acc1 = __builtin_amdgcn_wmma_f32_16x16x32_bf16(
                       false, ah[f], false, bl1, (short)0, acc1, false, false);
            acc0 = __builtin_amdgcn_wmma_f32_16x16x32_bf16(
                       false, al[f], false, bh0, (short)0, acc0, false, false);
            acc1 = __builtin_amdgcn_wmma_f32_16x16x32_bf16(
                       false, al[f], false, bh1, (short)0, acc1, false, false);
        }
        // hint the next iteration's column tiles (global_prefetch_b8)
        if (colBase + 32 < colEnd) {
            __builtin_prefetch(hi + (colBase + 32 + lane15) * DN, 0, 3);
            __builtin_prefetch(lo + (colBase + 32 + lane15) * DN, 0, 3);
            __builtin_prefetch(hi + (colBase + 48 + lane15) * DN, 0, 3);
            __builtin_prefetch(lo + (colBase + 48 + lane15) * DN, 0, 3);
        }
        // epilogue: exp + masked accumulation (co-executes with XDL WMMA)
#pragma unroll
        for (int r = 0; r < 8; ++r) {
            const int  rowG  = rowBase + r + laneHi * 8;
            const float e0   = __expf(acc0[r] * INV_TEMP);
            const float e1   = __expf(acc1[r] * INV_TEMP);
            const bool same0 = (cl0 == rl[r]);
            const bool same1 = (cl1 == rl[r]);
            pacc[r] += (same0 && (rowG != col0)) ? e0 : 0.f;
            nacc[r] += (!same0) ? e0 : 0.f;
            pacc[r] += (same1 && (rowG != col1)) ? e1 : 0.f;
            nacc[r] += (!same1) ? e1 : 0.f;
        }
    }

    // reduce across the 16 lanes of each half (xor masks stay in-half)
#pragma unroll
    for (int r = 0; r < 8; ++r) {
        float p = pacc[r], n = nacc[r];
#pragma unroll
        for (int m = 1; m < 16; m <<= 1) {
            p += __shfl_xor(p, m, 32);
            n += __shfl_xor(n, m, 32);
        }
        if (lane15 == r) {
            const int rowG = rowBase + r + laneHi * 8;
            atomicAdd(posOut + rowG, p);
            atomicAdd(negOut + rowG, n);
        }
    }
}

// ======================================================================
// Kernel 3: per-row loss assembly + global mean reduction into d_out.
// ======================================================================
__global__ void maccl_final_kernel(const int*   __restrict__ labels,
                                   const float* __restrict__ runSigma,
                                   const float* __restrict__ distSq,
                                   const float* __restrict__ pos,
                                   const float* __restrict__ neg,
                                   const float* __restrict__ scal,
                                   float* __restrict__ out) {
    const int i    = blockIdx.x * 256 + threadIdx.x;
    const int lane = threadIdx.x & 31;
    const int w    = threadIdx.x >> 5;
    __shared__ float sb[8];

    const float sum   = scal[0];
    const float sumsq = scal[1];
    const float cnt   = scal[2];
    const float nEl   = cnt * (float)DN;
    const float mean  = sum / nEl;
    const float var   = (sumsq - sum * mean) / (nEl - 1.0f);
    const float sigma = 0.9f * runSigma[0] + 0.1f * sqrtf(var);
    const float mAdap = 0.5f + 0.3f * sigma + 0.3f * (1.0f - 224.0f / 900.0f);

    const int   lab = labels[i];
    const float d2  = distSq[i];
    const float rc  = (lab == 0) ? d2 : 0.f;
    const float rm  = (lab == 1) ? fmaxf(mAdap - sqrtf(d2), 0.f) : 0.f;

    const float nSame  = (lab == 0) ? cnt : ((float)BN - cnt);
    const bool  hasB   = (nSame - 1.f > 0.f) && (((float)BN - nSame) > 0.f);
    const float p      = pos[i];
    const float n      = neg[i];
    const float rcon   = hasB ? -logf(p / (p + n + 1e-8f)) : 0.f;

    float raw = rc + rm + 0.5f * rcon;
#pragma unroll
    for (int o = 16; o > 0; o >>= 1) raw += __shfl_down(raw, o, 32);
    if (lane == 0) sb[w] = raw;
    __syncthreads();
    if (threadIdx.x == 0) {
        float t = 0.f;
#pragma unroll
        for (int k = 0; k < 8; ++k) t += sb[k];
        atomicAdd(out, t * (1.0f / (float)BN));
    }
}

// ======================================================================
// launch
// ======================================================================
extern "C" void kernel_launch(void* const* d_in, const int* in_sizes, int n_in,
                              void* d_out, int out_size, void* d_ws, size_t ws_size,
                              hipStream_t stream) {
    const float* feats   = (const float*)d_in[0];
    const int*   labels  = (const int*)d_in[1];
    const float* center  = (const float*)d_in[2];
    const float* runSig  = (const float*)d_in[3];
    float*       out     = (float*)d_out;

    char* ws = (char*)d_ws;
    const size_t planeBytes = (size_t)BN * DN * sizeof(__bf16); // 4 MB
    __bf16* hi     = (__bf16*)(ws);
    __bf16* lo     = (__bf16*)(ws + planeBytes);
    float*  distSq = (float*)(ws + 2 * planeBytes);
    float*  pos    = (float*)(ws + 2 * planeBytes + (size_t)BN * 4);
    float*  neg    = (float*)(ws + 2 * planeBytes + (size_t)BN * 8);
    float*  scal   = (float*)(ws + 2 * planeBytes + (size_t)BN * 12);

    maccl_zero_kernel <<<BN/256, 256, 0, stream>>>(scal, pos, neg, out);
    maccl_prep_kernel <<<BN,     256, 0, stream>>>(feats, labels, center, hi, lo,
                                                   distSq, scal);
    dim3 gGrid(BN / 128, NSPLIT);
    maccl_gram_kernel <<<gGrid,  256, 0, stream>>>(hi, lo, labels, pos, neg);
    maccl_final_kernel<<<BN/256, 256, 0, stream>>>(labels, runSig, distSq, pos,
                                                   neg, scal, out);
}
